// NTXentLoss_46016279609996
// MI455X (gfx1250) — matrix operand order repaired and checked by reading, not compiled, not run on hardware
//
#include <hip/hip_runtime.h>
#include <hip/hip_bf16.h>

typedef __attribute__((ext_vector_type(16))) _Float16 v16h;
typedef __attribute__((ext_vector_type(8)))  _Float16 v8h;
typedef __attribute__((ext_vector_type(8)))  float    v8f;
typedef __attribute__((ext_vector_type(4)))  float    v4f;

#define NPOS   4096          // N
#define NCOL   512           // E
#define NROWS  8192          // 2N
#define NSLAB  256           // NROWS / 32  (32-row i-slabs)
#define JCHUNK 8             // grid.y of simtile kernel
#define INVT   10.0f         // 1 / temperature
#define CEPS   1e-8f

// Workspace layout (bytes):
//   znh  : normalized rows as f16, NROWS x NCOL            = 8,388,608 B
//   pos  : positives, NPOS floats                          =    16,384 B
//   part : per-jchunk partial denominators, JCHUNK x NROWS =   262,144 B
#define ZNH_BYTES  ((size_t)NROWS * NCOL * 2)
#define POS_OFF    ZNH_BYTES
#define PART_OFF   (POS_OFF + (size_t)NPOS * 4)

// ---------------------------------------------------------------------------
// Stage 1: row-normalize z = concat(tab, ts) into f16. One wave per row.
// ---------------------------------------------------------------------------
__global__ void nx_normalize(const float* __restrict__ tab,
                             const float* __restrict__ ts,
                             _Float16* __restrict__ znh) {
  const int row  = (blockIdx.x * blockDim.x + threadIdx.x) >> 5;
  const int lane = threadIdx.x & 31;
  const float* src = (row < NPOS) ? (tab + (size_t)row * NCOL)
                                  : (ts + (size_t)(row - NPOS) * NCOL);
  const v4f* sp = (const v4f*)(src + lane * 16);
  v4f x0 = sp[0], x1 = sp[1], x2 = sp[2], x3 = sp[3];
  float s = 0.0f;
#pragma unroll
  for (int c = 0; c < 4; ++c)
    s += x0[c]*x0[c] + x1[c]*x1[c] + x2[c]*x2[c] + x3[c]*x3[c];
#pragma unroll
  for (int m = 16; m >= 1; m >>= 1) s += __shfl_xor(s, m, 32);
  const float inv = 1.0f / fmaxf(sqrtf(s), CEPS);

  v8h h0, h1;
#pragma unroll
  for (int c = 0; c < 4; ++c) {
    h0[c]     = (_Float16)(x0[c] * inv);
    h0[c + 4] = (_Float16)(x1[c] * inv);
    h1[c]     = (_Float16)(x2[c] * inv);
    h1[c + 4] = (_Float16)(x3[c] * inv);
  }
  v8h* op = (v8h*)(znh + (size_t)row * NCOL + lane * 16);
  op[0] = h0;
  op[1] = h1;
}

// ---------------------------------------------------------------------------
// Stage 2: positives pos[i] = dot(tab_i, ts_i)/(max(|tab_i|,eps)*max(|ts_i|,eps))
// Exact f32 math matching the reference. One wave per i, single plain store.
// ---------------------------------------------------------------------------
__global__ void nx_positives(const float* __restrict__ tab,
                             const float* __restrict__ ts,
                             float* __restrict__ pos) {
  const int i    = (blockIdx.x * blockDim.x + threadIdx.x) >> 5;
  const int lane = threadIdx.x & 31;
  const v4f* ap = (const v4f*)(tab + (size_t)i * NCOL + lane * 16);
  const v4f* bp = (const v4f*)(ts  + (size_t)i * NCOL + lane * 16);
  float dot = 0.0f, na = 0.0f, nb = 0.0f;
#pragma unroll
  for (int k = 0; k < 4; ++k) {
    v4f a = ap[k], b = bp[k];
#pragma unroll
    for (int c = 0; c < 4; ++c) {
      dot += a[c] * b[c];
      na  += a[c] * a[c];
      nb  += b[c] * b[c];
    }
  }
#pragma unroll
  for (int m = 16; m >= 1; m >>= 1) {
    dot += __shfl_xor(dot, m, 32);
    na  += __shfl_xor(na,  m, 32);
    nb  += __shfl_xor(nb,  m, 32);
  }
  if (lane == 0)
    pos[i] = dot / (fmaxf(sqrtf(na), CEPS) * fmaxf(sqrtf(nb), CEPS));
}

// ---------------------------------------------------------------------------
// Stage 3: fused tile GEMM + exp + masked row-sum.  M=32 per wave.
// Block: 128 threads (4 waves). grid = (NSLAB, JCHUNK).
// Each wave owns a 32-row A slab (two register fragment sets, 2x128 VGPRs)
// and runs TWO WMMAs per B fragment -> 32 FLOP per loaded byte.
// A fragments (ISA 16-bit A 16x32 layout): lane g=L>>4, r=L&15 holds row r,
//   halves[0..7] = K g*8..g*8+7, halves[8..15] = K 16+g*8..16+g*8+7.
// B fragments (32x16): lane holds column r = L&15, halves[0..7] = K g*8..g*8+7,
//   halves[8..15] = K 16+g*8..16+g*8+7 of row (jt*16+r) of znh.
// ---------------------------------------------------------------------------
__global__ __launch_bounds__(128, 1) void nx_simtile(const _Float16* __restrict__ znh,
                                                     float* __restrict__ part) {
  const int it     = blockIdx.x;          // 32-row slab index
  const int chunk  = blockIdx.y;
  const int w      = threadIdx.x >> 5;
  const int lane   = threadIdx.x & 31;
  const int l15    = lane & 15;
  const int g      = lane >> 4;

  // Preload A fragments for both 16-row tiles of the slab (K=512 -> 16 steps).
  const _Float16* arow0 = znh + (size_t)(it * 32 + l15) * NCOL;
  const _Float16* arow1 = znh + (size_t)(it * 32 + 16 + l15) * NCOL;
  v16h a0[16], a1[16];
#pragma unroll
  for (int k = 0; k < 16; ++k) {
    const v8h* ap = (const v8h*)(arow0 + k * 32 + g * 8);
    v8h lo = ap[0];
    v8h hi = ap[2];
    a0[k] = __builtin_shufflevector(lo, hi, 0,1,2,3,4,5,6,7,
                                            8,9,10,11,12,13,14,15);
  }
#pragma unroll
  for (int k = 0; k < 16; ++k) {
    const v8h* ap = (const v8h*)(arow1 + k * 32 + g * 8);
    v8h lo = ap[0];
    v8h hi = ap[2];
    a1[k] = __builtin_shufflevector(lo, hi, 0,1,2,3,4,5,6,7,
                                            8,9,10,11,12,13,14,15);
  }

  const float SCL = 1.44269504088896340736f * INVT;  // exp(x/T) = exp2(x*SCL)
  const int row0 = it * 32 + g * 8;        // rows of acc0 set
  const int row1 = row0 + 16;              // rows of acc1 set

  float acc0[8], acc1[8];
#pragma unroll
  for (int v = 0; v < 8; ++v) { acc0[v] = 0.0f; acc1[v] = 0.0f; }

  const int jt0 = chunk * 64 + w * 16;
  for (int jj = 0; jj < 16; ++jj) {
    const int jt = jt0 + jj;
    const _Float16* brow = znh + (size_t)(jt * 16 + l15) * NCOL + g * 8;
    v8f c0 = {}, c1 = {};
#pragma unroll
    for (int k = 0; k < 16; ++k) {
      const v8h* bp = (const v8h*)(brow + k * 32);
      v16h b = __builtin_shufflevector(bp[0], bp[2], 0,1,2,3,4,5,6,7,
                                       8,9,10,11,12,13,14,15);
      c0 = __builtin_amdgcn_wmma_f32_16x16x32_f16(
               false, a0[k], false, b, (short)0, c0, false, false);
      c1 = __builtin_amdgcn_wmma_f32_16x16x32_f16(
               false, a1[k], false, b, (short)0, c1, false, false);
    }
    const int colg = jt * 16 + l15;
#pragma unroll
    for (int v = 0; v < 8; ++v) {
      float e0 = exp2f(c0[v] * SCL);
      float e1 = exp2f(c1[v] * SCL);
      if (row0 + v != colg) acc0[v] += e0;   // pairwise_mask: drop diagonal
      if (row1 + v != colg) acc1[v] += e1;
    }
  }

  // Reduce across the 16 lanes of each half (same rows, different columns).
#pragma unroll
  for (int v = 0; v < 8; ++v) {
    float a = acc0[v];
    a += __shfl_xor(a, 1, 32);
    a += __shfl_xor(a, 2, 32);
    a += __shfl_xor(a, 4, 32);
    a += __shfl_xor(a, 8, 32);
    acc0[v] = a;
    float b = acc1[v];
    b += __shfl_xor(b, 1, 32);
    b += __shfl_xor(b, 2, 32);
    b += __shfl_xor(b, 4, 32);
    b += __shfl_xor(b, 8, 32);
    acc1[v] = b;
  }

  // Deterministic combine of the 4 waves via LDS, single store per row.
  __shared__ float sden[4 * 32];
  if (l15 == 0) {
#pragma unroll
    for (int v = 0; v < 8; ++v) {
      sden[w * 32 + g * 8 + v]      = acc0[v];
      sden[w * 32 + 16 + g * 8 + v] = acc1[v];
    }
  }
  __syncthreads();
  if (threadIdx.x < 32) {
    const int r = threadIdx.x;
    float s = sden[r] + sden[32 + r] + sden[64 + r] + sden[96 + r];
    part[(size_t)chunk * NROWS + it * 32 + r] = s;
  }
}

// ---------------------------------------------------------------------------
// Stage 4: loss = (sum_r log(denom_r) - 2*invT*sum_i pos_i) / (2N)
// ---------------------------------------------------------------------------
__global__ void nx_finalize(const float* __restrict__ part,
                            const float* __restrict__ pos,
                            float* __restrict__ out) {
  __shared__ float red[256];
  float s = 0.0f;
  for (int r = threadIdx.x; r < NROWS; r += 256) {
    float d = 0.0f;
#pragma unroll
    for (int y = 0; y < JCHUNK; ++y) d += part[(size_t)y * NROWS + r];
    s += __logf(d);
  }
  for (int i = threadIdx.x; i < NPOS; i += 256)
    s -= 2.0f * INVT * pos[i];
  red[threadIdx.x] = s;
  __syncthreads();
  for (int off = 128; off > 0; off >>= 1) {
    if (threadIdx.x < off) red[threadIdx.x] += red[threadIdx.x + off];
    __syncthreads();
  }
  if (threadIdx.x == 0) out[0] = red[0] / (float)NROWS;
}

// ---------------------------------------------------------------------------
extern "C" void kernel_launch(void* const* d_in, const int* in_sizes, int n_in,
                              void* d_out, int out_size, void* d_ws, size_t ws_size,
                              hipStream_t stream) {
  const float* tab = (const float*)d_in[0];
  const float* ts  = (const float*)d_in[1];
  float* out = (float*)d_out;

  char* ws = (char*)d_ws;
  _Float16* znh = (_Float16*)(ws);
  float*    pos  = (float*)(ws + POS_OFF);
  float*    part = (float*)(ws + PART_OFF);

  // Stage 1: 8192 rows, one wave each -> 262144 threads.
  nx_normalize<<<dim3(NROWS * 32 / 256), dim3(256), 0, stream>>>(tab, ts, znh);
  // Stage 2: 4096 pairs, one wave each.
  nx_positives<<<dim3(NPOS * 32 / 256), dim3(256), 0, stream>>>(tab, ts, pos);
  // Stage 3: 256 32-row i-slabs x 8 j-chunks, 4 waves per block.
  nx_simtile<<<dim3(NSLAB, JCHUNK), dim3(128), 0, stream>>>(znh, part);
  // Stage 4: single block reduction.
  nx_finalize<<<dim3(1), dim3(256), 0, stream>>>(part, pos, out);
}
